// TopoLoss_78821239816704
// MI455X (gfx1250) — matrix operand order repaired and checked by reading, not compiled, not run on hardware
//
#include <hip/hip_runtime.h>
#include <hip/hip_bf16.h>
#include <math.h>

// ---------------------------------------------------------------------------
// CDNA5 (gfx1250) wave32 WMMA types
// ---------------------------------------------------------------------------
typedef __attribute__((ext_vector_type(16))) __bf16        v16bf;
typedef __attribute__((ext_vector_type(8)))  float         v8f;
typedef __attribute__((ext_vector_type(4)))  float         f32x4;
typedef __attribute__((ext_vector_type(4)))  unsigned int  u32x4;

union BF16x16 {
    v16bf v;
    unsigned int u[8];
    u32x4 q[2];
};

__device__ __forceinline__ v8f wmma_bf16(v16bf a, v16bf b, v8f c) {
    // v_wmma_f32_16x16x32_bf16  D = A(16x32) * B(32x16) + C
    return __builtin_amdgcn_wmma_f32_16x16x32_bf16(
        /*neg_a=*/false, a, /*neg_b=*/false, b,
        /*c_mod=*/(short)0, c, /*reuse_a=*/false, /*reuse_b=*/false);
}

// f32 pair -> packed bf16 (truncation) in ONE v_perm_b32.
__device__ __forceinline__ unsigned int pack_bf16(float lo, float hi) {
    return __builtin_amdgcn_perm(__float_as_uint(hi), __float_as_uint(lo),
                                 0x07060302u);
}

#define NN     8192
#define KK     64
#define CARD   64
#define BIGF   1e9f
#define KSPLIT 4
#define KCH    (NN / KSPLIT)   // 2048 reduction elems per k-split wave

// ---------------------------------------------------------------------------
// Kernel 0: St[c][n] (bf16, 64 x 8192) = S[n][c] (f32)  — transpose+convert
// ---------------------------------------------------------------------------
__global__ __launch_bounds__(256) void k_transS(const float* __restrict__ S,
                                                unsigned short* __restrict__ St) {
    int tid = blockIdx.x * 256 + threadIdx.x;     // 0 .. 262143
    int c  = tid >> 12;                           // 0..63
    int n2 = tid & 4095;                          // pair of n
    int n  = n2 * 2;
    float a = S[(size_t)n * KK + c];
    float b = S[(size_t)(n + 1) * KK + c];
    ((unsigned int*)St)[(size_t)c * 4096 + n2] = pack_bf16(a, b);
}

// ---------------------------------------------------------------------------
// Kernel 1: T = adj @ S via WMMA, 2048 waves (split-K x4 for HBM concurrency).
// Explicit ping-pong double buffer (s0/s1, unroll-2) so the compiler emits no
// stage/acc register copies and no rotated-loop guards; loads are issued two
// iterations ahead of consumption. adj loads are non-temporal (read-once
// 256MB stream), with device-scope prefetch ahead of the stream.
// The 4 K-split waves of a row tile share a block and reduce via LDS;
// wave ks==0 writes the bf16 transposed tile Tt[c][n] = T[n][c].
// ---------------------------------------------------------------------------
struct Stage {
    f32x4 a0, a1, a2, a3;   // 16 f32 of adj (A fragment, this lane)
    u32x4 b[8];             // 4 B fragments (bf16, 2 x 16B each)
};

__device__ __forceinline__ void stage_load(Stage& s, const float* __restrict__ arowp,
                                           const unsigned short* __restrict__ St,
                                           int n0, int l15, int hl) {
    const float* p = arowp + n0 + 8 * hl;
    s.a0 = __builtin_nontemporal_load((const f32x4*)(p));
    s.a1 = __builtin_nontemporal_load((const f32x4*)(p + 4));
    s.a2 = __builtin_nontemporal_load((const f32x4*)(p + 16));
    s.a3 = __builtin_nontemporal_load((const f32x4*)(p + 20));
#pragma unroll
    for (int ct = 0; ct < 4; ++ct) {
        const u32x4* bp = (const u32x4*)(St + (size_t)(ct * 16 + l15) * NN
                                            + n0 + 8 * hl);
        s.b[2 * ct]     = bp[0];
        s.b[2 * ct + 1] = bp[2];
    }
}

__device__ __forceinline__ void stage_mma(const Stage& s, v8f acc[4]) {
    BF16x16 a;
    a.u[0] = pack_bf16(s.a0.x, s.a0.y);  a.u[1] = pack_bf16(s.a0.z, s.a0.w);
    a.u[2] = pack_bf16(s.a1.x, s.a1.y);  a.u[3] = pack_bf16(s.a1.z, s.a1.w);
    a.u[4] = pack_bf16(s.a2.x, s.a2.y);  a.u[5] = pack_bf16(s.a2.z, s.a2.w);
    a.u[6] = pack_bf16(s.a3.x, s.a3.y);  a.u[7] = pack_bf16(s.a3.z, s.a3.w);
#pragma unroll
    for (int ct = 0; ct < 4; ++ct) {
        BF16x16 b;
        b.q[0] = s.b[2 * ct];
        b.q[1] = s.b[2 * ct + 1];
        acc[ct] = wmma_bf16(a.v, b.v, acc[ct]);
    }
}

__global__ __launch_bounds__(256) void k_gemm1(const float* __restrict__ adj,
                                               const unsigned short* __restrict__ St,
                                               unsigned short* __restrict__ Tt) {
    __shared__ float lds[8 * 1024];               // 8 waves x (4 frag x 8 x 32)
    const int lane = threadIdx.x & 31;
    const int wave = threadIdx.x >> 5;
    const int rtl  = wave >> 2;                   // row tile within block (0..1)
    const int ks   = wave & 3;                    // k-split id
    const int rt   = blockIdx.x * 2 + rtl;        // 0..511
    const int row0 = rt * 16;
    const int hl   = lane >> 4;
    const int l15  = lane & 15;
    const int nbeg = ks * KCH;
    const int nend = nbeg + KCH;

    v8f acc[4];
#pragma unroll
    for (int t = 0; t < 4; ++t) acc[t] = (v8f){0,0,0,0,0,0,0,0};

    const float* arowp = adj + (size_t)(row0 + l15) * NN;

    // ping-pong double buffer, pipeline depth 2, no register copies
    Stage s0, s1;
    stage_load(s0, arowp, St, nbeg,      l15, hl);
    stage_load(s1, arowp, St, nbeg + 32, l15, hl);
    for (int n0 = nbeg + 64; n0 < nend; n0 += 64) {
        stage_mma(s0, acc);
        stage_load(s0, arowp, St, n0, l15, hl);
        __builtin_prefetch(arowp + n0 + 8 * hl + 2048, 0, 1);  // ~64 iters ahead
        stage_mma(s1, acc);
        stage_load(s1, arowp, St, n0 + 32, l15, hl);
    }
    stage_mma(s0, acc);
    stage_mma(s1, acc);

    // cross-wave (k-split) reduction through LDS, conflict-free layout
    float* lw = lds + wave * 1024;
#pragma unroll
    for (int ct = 0; ct < 4; ++ct)
#pragma unroll
        for (int r = 0; r < 8; ++r)
            lw[ct * 256 + r * 32 + lane] = acc[ct][r];
    __syncthreads();

    if (ks == 0) {
#pragma unroll
        for (int ct = 0; ct < 4; ++ct) {
            float s[8];
#pragma unroll
            for (int r = 0; r < 8; ++r) {
                float t = 0.0f;
#pragma unroll
                for (int j = 0; j < 4; ++j)
                    t += lds[(rtl * 4 + j) * 1024 + ct * 256 + r * 32 + lane];
                s[r] = t;
            }
            int c = ct * 16 + l15;
            u32x4 o;
            o.x = pack_bf16(s[0], s[1]);
            o.y = pack_bf16(s[2], s[3]);
            o.z = pack_bf16(s[4], s[5]);
            o.w = pack_bf16(s[6], s[7]);
            *(u32x4*)(Tt + (size_t)c * NN + row0 + 8 * hl) = o;
        }
    }
}

// ---------------------------------------------------------------------------
// Kernel 2: C = S^T @ T (64x64) via WMMA. 16 waves, one 16x16 tile each,
// 4 independent accumulators to break the WMMA dependency chain.
// ---------------------------------------------------------------------------
__global__ __launch_bounds__(512) void k_gemm2(const unsigned short* __restrict__ St,
                                               const unsigned short* __restrict__ Tt,
                                               float* __restrict__ C) {
    const int lane = threadIdx.x & 31;
    const int wave = threadIdx.x >> 5;            // 0..15
    const int a0 = (wave >> 2) * 16;
    const int c0 = (wave & 3) * 16;
    const int hl = lane >> 4;
    const int l15 = lane & 15;

    v8f acc[4];
#pragma unroll
    for (int t = 0; t < 4; ++t) acc[t] = (v8f){0,0,0,0,0,0,0,0};

    const unsigned short* arow = St + (size_t)(a0 + l15) * NN;   // S^T row a
    const unsigned short* brow = Tt + (size_t)(c0 + l15) * NN;   // T col c

    for (int n0 = 0; n0 < NN; n0 += 128) {
#pragma unroll
        for (int u = 0; u < 4; ++u) {
            int n = n0 + u * 32;
            BF16x16 a, b;
            a.q[0] = *(const u32x4*)(arow + n + 8 * hl);
            a.q[1] = *(const u32x4*)(arow + n + 16 + 8 * hl);
            const u32x4* bp = (const u32x4*)(brow + n + 8 * hl);
            b.q[0] = bp[0];
            b.q[1] = bp[2];
            acc[u] = wmma_bf16(a.v, b.v, acc[u]);
        }
    }
    v8f accf = (acc[0] + acc[1]) + (acc[2] + acc[3]);
#pragma unroll
    for (int r = 0; r < 8; ++r)
        C[(size_t)(a0 + r + 8 * hl) * KK + c0 + l15] = accf[r];
}

// ---------------------------------------------------------------------------
// Kernel 3: symmetrize, normalize by max, D = 1 - C/max, zero diagonal
// ---------------------------------------------------------------------------
__global__ __launch_bounds__(256) void k_finalD(const float* __restrict__ C,
                                                float* __restrict__ D) {
    __shared__ float red[256];
    int t = threadIdx.x;
    float mx = -1e30f;
    for (int e = t; e < 4096; e += 256) {
        int i = e >> 6, j = e & 63;
        float cs = 0.5f * (C[e] + C[j * 64 + i]);
        mx = fmaxf(mx, cs);
    }
    red[t] = mx; __syncthreads();
    for (int s = 128; s > 0; s >>= 1) {
        if (t < s) red[t] = fmaxf(red[t], red[t + s]);
        __syncthreads();
    }
    float inv = 1.0f / (red[0] + 1e-12f);
    for (int e = t; e < 4096; e += 256) {
        int i = e >> 6, j = e & 63;
        float cs = 0.5f * (C[e] + C[j * 64 + i]);
        float d = 1.0f - cs * inv;
        if (i == j) d = 0.0f;
        D[e] = d;
    }
}

// ---------------------------------------------------------------------------
// Kernel 4: Prim's MST over 64 nodes + dim-0 diagram (stable descending sort)
// ---------------------------------------------------------------------------
__global__ __launch_bounds__(64) void k_prim(const float* __restrict__ D,
                                             int* __restrict__ mst,
                                             float* __restrict__ dgm0) {
    __shared__ float key[64];
    __shared__ int   par[64];
    __shared__ int   intree[64];
    __shared__ float rv[64];
    __shared__ int   ri[64];
    __shared__ float ws[64];
    int v = threadIdx.x;

    for (int e = v; e < 4096; e += 64) mst[e] = 0;
    key[v]    = D[v];          // D[0][v]
    par[v]    = 0;
    intree[v] = (v == 0) ? 1 : 0;
    __syncthreads();

    for (int step = 0; step < 63; ++step) {
        rv[v] = intree[v] ? BIGF : key[v];
        ri[v] = v;
        __syncthreads();
        for (int s = 32; s > 0; s >>= 1) {
            if (v < s && rv[v + s] < rv[v]) { rv[v] = rv[v + s]; ri[v] = ri[v + s]; }
            __syncthreads();
        }
        int vstar = ri[0];
        if (v == 0) {
            ws[step] = rv[0];
            int u = par[vstar];
            mst[u * 64 + vstar] = 1;
            mst[vstar * 64 + u] = 1;
            intree[vstar] = 1;
        }
        __syncthreads();
        float cand = D[vstar * 64 + v];
        if (!intree[v] && cand < key[v]) { key[v] = cand; par[v] = vstar; }
        __syncthreads();
    }
    if (v == 63) ws[63] = 0.0f;   // pad to CARD
    __syncthreads();

    // stable descending rank-sort -> deaths; births are 0
    float x = ws[v];
    int rank = 0;
    for (int j = 0; j < 64; ++j) {
        float y = ws[j];
        if (y > x || (y == x && j < v)) rank++;
    }
    dgm0[rank * 2 + 0] = 0.0f;
    dgm0[rank * 2 + 1] = x;
}

// ---------------------------------------------------------------------------
// Kernel 5: dim-1 deaths + persistence with validity mask
// ---------------------------------------------------------------------------
__global__ __launch_bounds__(256) void k_death(const float* __restrict__ D,
                                               const int* __restrict__ mst,
                                               float* __restrict__ pers,
                                               float* __restrict__ death) {
    int e = blockIdx.x * 256 + threadIdx.x;       // 0..4095
    int i = e >> 6, j = e & 63;
    float dij = D[e];
    float mn = BIGF;
    for (int k = 0; k < 64; ++k) {
        if (k == i || k == j) continue;
        float m = fmaxf(dij, fmaxf(D[i * 64 + k], D[j * 64 + k]));
        mn = fminf(mn, m);
    }
    death[e] = mn;
    bool valid = (j > i) && (mst[e] == 0);
    pers[e] = valid ? (mn - dij) : -1.0f;
}

// ---------------------------------------------------------------------------
// Kernel 6: stable top-64 by persistence (counting rank == jax top_k order)
// ---------------------------------------------------------------------------
__global__ __launch_bounds__(256) void k_topk(const float* __restrict__ D,
                                              const float* __restrict__ pers,
                                              const float* __restrict__ death,
                                              float* __restrict__ dgm1) {
    int e = blockIdx.x * 256 + threadIdx.x;
    float x = pers[e];
    int rank = 0;
    for (int j = 0; j < 4096; ++j) {
        float y = pers[j];
        if (y > x || (y == x && j < e)) { if (++rank >= CARD) break; }
    }
    if (rank < CARD) {
        bool keep = x > 0.0f;
        dgm1[rank * 2 + 0] = keep ? D[e]     : 0.0f;
        dgm1[rank * 2 + 1] = keep ? death[e] : 0.0f;
    }
}

// ---------------------------------------------------------------------------
// Kernel 7: entropic Sinkhorn W1; 4 independent blocks, cost matrix in LDS
// ---------------------------------------------------------------------------
__global__ __launch_bounds__(128) void k_sinkhorn(const float* __restrict__ dgm0,
                                                  const float* __restrict__ dgm1,
                                                  const float* __restrict__ gb0,
                                                  const float* __restrict__ gb1,
                                                  const float* __restrict__ gn0,
                                                  const float* __restrict__ gn1,
                                                  float* __restrict__ partial) {
    __shared__ float Cm[65][65];
    __shared__ float f[65], g[65], loga[65], logb[65];
    __shared__ float red[128];
    const float eps = 0.01f;
    int p = blockIdx.x;
    int t = threadIdx.x;
    const float* X = (p == 0 || p == 2) ? dgm0 : dgm1;
    const float* Y = (p == 0) ? gb0 : (p == 1) ? gb1 : (p == 2) ? gn0 : gn1;

    for (int e = t; e < 65 * 65; e += 128) {
        int i = e / 65, j = e % 65;
        float c;
        if (i < 64 && j < 64)
            c = fmaxf(fabsf(X[i * 2] - Y[j * 2]), fabsf(X[i * 2 + 1] - Y[j * 2 + 1]));
        else if (i < 64) c = 0.5f * (X[i * 2 + 1] - X[i * 2]);
        else if (j < 64) c = 0.5f * (Y[j * 2 + 1] - Y[j * 2]);
        else             c = 0.0f;
        Cm[i][j] = c;
    }
    if (t < 65) {
        f[t] = 0.0f; g[t] = 0.0f;
        loga[t] = (t < 64) ? 0.0f : logf(64.0f);
        logb[t] = loga[t];
    }
    __syncthreads();

    for (int it = 0; it < 300; ++it) {
        if (t < 65) {
            float m = -1e30f;
            for (int j = 0; j < 65; ++j) m = fmaxf(m, g[j] - Cm[t][j]);
            float s = 0.0f;
            for (int j = 0; j < 65; ++j) s += expf((g[j] - Cm[t][j] - m) / eps);
            f[t] = eps * loga[t] - m - eps * logf(s);
        }
        __syncthreads();
        if (t < 65) {
            float m = -1e30f;
            for (int i = 0; i < 65; ++i) m = fmaxf(m, f[i] - Cm[i][t]);
            float s = 0.0f;
            for (int i = 0; i < 65; ++i) s += expf((f[i] - Cm[i][t] - m) / eps);
            g[t] = eps * logb[t] - m - eps * logf(s);
        }
        __syncthreads();
    }

    float acc = 0.0f;
    for (int e = t; e < 65 * 65; e += 128) {
        int i = e / 65, j = e % 65;
        float c = Cm[i][j];
        acc += expf((f[i] + g[j] - c) / eps) * c;
    }
    red[t] = acc; __syncthreads();
    for (int s = 64; s > 0; s >>= 1) {
        if (t < s) red[t] += red[t + s];
        __syncthreads();
    }
    if (t == 0) partial[p] = red[0];
}

// ---------------------------------------------------------------------------
// Kernel 8: final scalar (fixed summation order -> deterministic)
// ---------------------------------------------------------------------------
__global__ void k_final(const float* __restrict__ partial, float* __restrict__ out) {
    if (threadIdx.x == 0 && blockIdx.x == 0)
        out[0] = 0.1f * (((partial[0] + partial[1]) + partial[2]) + partial[3]);
}

// ---------------------------------------------------------------------------
extern "C" void kernel_launch(void* const* d_in, const int* in_sizes, int n_in,
                              void* d_out, int out_size, void* d_ws, size_t ws_size,
                              hipStream_t stream) {
    (void)in_sizes; (void)n_in; (void)out_size; (void)ws_size;

    const float* adj = (const float*)d_in[0];
    const float* S   = (const float*)d_in[1];
    const float* gb0 = (const float*)d_in[2];
    const float* gb1 = (const float*)d_in[3];
    const float* gn0 = (const float*)d_in[4];
    const float* gn1 = (const float*)d_in[5];
    float* out = (float*)d_out;

    // scratch layout (~2.2 MB)
    char* ws = (char*)d_ws;
    unsigned short* St = (unsigned short*)(ws);                 // 1 MB
    unsigned short* Tt = (unsigned short*)(ws + (1u << 20));    // 1 MB
    float* Cmat    = (float*)(ws + (2u << 20));                 // 16 KB
    float* Dmat    = Cmat + 4096;                               // 16 KB
    int*   mst     = (int*)(Dmat + 4096);                       // 16 KB
    float* dgm0    = (float*)(mst + 4096);                      // 512 B
    float* dgm1    = dgm0 + 128;                                // 512 B
    float* pers    = dgm1 + 128;                                // 16 KB
    float* death   = pers + 4096;                               // 16 KB
    float* partial = death + 4096;                              // 16 B

    k_transS  <<<1024, 256, 0, stream>>>(S, St);
    k_gemm1   <<<256,  256, 0, stream>>>(adj, St, Tt);   // 2048 waves, split-K
    k_gemm2   <<<1,    512, 0, stream>>>(St, Tt, Cmat);
    k_finalD  <<<1,    256, 0, stream>>>(Cmat, Dmat);
    k_prim    <<<1,     64, 0, stream>>>(Dmat, mst, dgm0);
    k_death   <<<16,   256, 0, stream>>>(Dmat, mst, pers, death);
    k_topk    <<<16,   256, 0, stream>>>(Dmat, pers, death, dgm1);
    k_sinkhorn<<<4,    128, 0, stream>>>(dgm0, dgm1, gb0, gb1, gn0, gn1, partial);
    k_final   <<<1,      1, 0, stream>>>(partial, out);
}
